// SalientTransformer_84232898609397
// MI455X (gfx1250) — matrix operand order repaired and checked by reading, not compiled, not run on hardware
//
#include <hip/hip_runtime.h>
#include <stdint.h>

#define DEV __device__ __forceinline__

constexpr int Bb = 4, Nn = 2048, Dd = 384, Hh = 6, HD = 64, KNN = 8;
constexpr int BN = Bb * Nn;              // 8192 tokens
constexpr int GF_ROWS = Bb * KNN * Nn;   // 65536 edge rows

typedef __attribute__((ext_vector_type(16))) __bf16 v16bf;
typedef __attribute__((ext_vector_type(8)))  float  v8f;

#if __has_builtin(__builtin_amdgcn_tensor_load_to_lds) && __has_builtin(__builtin_amdgcn_s_wait_tensorcnt)
#define HAVE_TDM 1
typedef __attribute__((ext_vector_type(4))) unsigned int u32x4;
typedef __attribute__((ext_vector_type(8))) int i32x8;
typedef __attribute__((ext_vector_type(4))) int i32x4;
#endif

union Frag {
  uint4 u4[2];
  v16bf v;
};

DEV unsigned short f2bf(float f) {
  unsigned int u = __float_as_uint(f);
  u += 0x7FFFu + ((u >> 16) & 1u);   // round-to-nearest-even
  return (unsigned short)(u >> 16);
}
DEV float bf2f(unsigned short h) { return __uint_as_float(((unsigned int)h) << 16); }

DEV v8f wmma_bf16(v16bf a, v16bf b, v8f c) {
  return __builtin_amdgcn_wmma_f32_16x16x32_bf16(false, a, false, b, (short)0, c,
                                                 false, false);
}

DEV unsigned int bfsub2(unsigned int a, unsigned int b) {
  float al = __uint_as_float(a << 16);
  float ah = __uint_as_float(a & 0xFFFF0000u);
  float bl = __uint_as_float(b << 16);
  float bh = __uint_as_float(b & 0xFFFF0000u);
  return (unsigned int)f2bf(al - bl) | ((unsigned int)f2bf(ah - bh) << 16);
}
DEV uint4 bfsub8(uint4 a, uint4 b) {
  uint4 r;
  r.x = bfsub2(a.x, b.x); r.y = bfsub2(a.y, b.y);
  r.z = bfsub2(a.z, b.z); r.w = bfsub2(a.w, b.w);
  return r;
}

DEV float gelu_exact(float v) { return 0.5f * v * (1.f + erff(v * 0.70710678118f)); }

DEV float blockReduceSum128(float v, float* red) {
  int t = threadIdx.x;
  red[t] = v; __syncthreads();
  for (int s = 64; s > 0; s >>= 1) { if (t < s) red[t] += red[t + s]; __syncthreads(); }
  float r = red[0]; __syncthreads();
  return r;
}

// ---------------------------------------------------------------- utilities
__global__ void cvt_transpose_kernel(const float* __restrict__ src,
                                     unsigned short* __restrict__ dst, int R, int C) {
  int i = blockIdx.x * blockDim.x + threadIdx.x;
  if (i >= R * C) return;
  int r = i / C, c = i - r * C;
  dst[(size_t)c * R + r] = f2bf(src[i]);   // [C][R] bf16, K-major for B-fragments
}

__global__ void fill_neg_kernel(float* __restrict__ p, int n) {
  int i = blockIdx.x * blockDim.x + threadIdx.x;
  if (i < n) p[i] = -1e30f;
}

__global__ void knn_to_gi_kernel(const float* __restrict__ knn_out,
                                 unsigned short* __restrict__ gi, int n) {
  int i = blockIdx.x * blockDim.x + threadIdx.x;
  if (i >= n) return;
  int row = i / Dd, d = i - row * Dd;
  gi[(size_t)row * (2 * Dd) + Dd + d] = f2bf(knn_out[i]);
}

// ---------------------------------------------------------------- LayerNorm 1
__global__ __launch_bounds__(128)
void ln1_kernel(const float* __restrict__ x, const float* __restrict__ g,
                const float* __restrict__ b, unsigned short* __restrict__ nx) {
  __shared__ float red[128];
  int row = blockIdx.x, t = threadIdx.x;
  const float* xr = x + (size_t)row * Dd;
  float v0 = xr[t], v1 = xr[t + 128], v2 = xr[t + 256];
  float mu = blockReduceSum128(v0 + v1 + v2, red) * (1.f / Dd);
  float d0 = v0 - mu, d1 = v1 - mu, d2 = v2 - mu;
  float var = blockReduceSum128(d0 * d0 + d1 * d1 + d2 * d2, red) * (1.f / Dd);
  float rstd = rsqrtf(var + 1e-5f);
  unsigned short* nr = nx + (size_t)row * Dd;
  nr[t]       = f2bf(d0 * rstd * g[t]       + b[t]);
  nr[t + 128] = f2bf(d1 * rstd * g[t + 128] + b[t + 128]);
  nr[t + 256] = f2bf(d2 * rstd * g[t + 256] + b[t + 256]);
}

// -------------------------------------------------- generic bf16 WMMA GEMM
// A[M][Kin] bf16 row-major, Bt[Nout][Kin] bf16 (pre-transposed weights).
// 256 thr = 8 waves, wave grid 4(M)x2(N), wave tile 32x64 -> block tile 128x128.
enum { EPI_QKV = 0, EPI_PROJ = 1, EPI_GELU_BF16 = 2, EPI_SIGMOID_F32 = 3, EPI_ADD_RES = 4 };

__global__ __launch_bounds__(256)
void gemm_bf16_kernel(int Kin, int Nout,
                      const unsigned short* __restrict__ A,
                      const unsigned short* __restrict__ Bt,
                      const float* __restrict__ bias, int epi,
                      float* __restrict__ outf,
                      unsigned short* __restrict__ outb,
                      const float* __restrict__ aux) {
  const int tid = threadIdx.x;
  const int w = tid >> 5, lane = tid & 31, hl = lane >> 4, l16 = lane & 15;
  const int wm = w >> 1, wn = w & 1;
  const int rowBase = blockIdx.x * 128 + wm * 32;
  const int colBase = blockIdx.y * 128 + wn * 64;

  v8f acc[2][4];
  #pragma unroll
  for (int i = 0; i < 2; ++i)
    #pragma unroll
    for (int j = 0; j < 4; ++j)
      #pragma unroll
      for (int e = 0; e < 8; ++e) acc[i][j][e] = 0.f;

  for (int kk = 0; kk < Kin; kk += 32) {
    Frag af[2], bf[4];
    #pragma unroll
    for (int mt = 0; mt < 2; ++mt) {
      int row = rowBase + mt * 16 + l16;
      int k0 = kk + hl * 8;   // A layout: VGPR0-3 = K kb..kb+7, VGPR4-7 = K kb+16..kb+23
      const unsigned short* p = A + (size_t)row * Kin + k0;
      af[mt].u4[0] = *reinterpret_cast<const uint4*>(p);
      af[mt].u4[1] = *reinterpret_cast<const uint4*>(p + 16);
    }
    #pragma unroll
    for (int nt = 0; nt < 4; ++nt) {
      int col = colBase + nt * 16 + l16;
      int k0 = kk + hl * 8;   // B layout: lane=N, K kb..kb+7 then kb+16..kb+23 (K-major Bt)
      const unsigned short* p = Bt + (size_t)col * Kin + k0;
      bf[nt].u4[0] = *reinterpret_cast<const uint4*>(p);
      bf[nt].u4[1] = *reinterpret_cast<const uint4*>(p + 16);
    }
    if (kk + 32 < Kin) {
      __builtin_prefetch(A + (size_t)(rowBase + l16) * Kin + kk + 32, 0, 3);
      __builtin_prefetch(Bt + (size_t)(colBase + l16) * Kin + kk + 32, 0, 3);
    }
    #pragma unroll
    for (int mt = 0; mt < 2; ++mt)
      #pragma unroll
      for (int nt = 0; nt < 4; ++nt)
        acc[mt][nt] = wmma_bf16(af[mt].v, bf[nt].v, acc[mt][nt]);
  }

  // epilogue: C/D layout -> VGPR r: rows r (lanes 0-15) / r+8 (lanes 16-31), col = l16
  const size_t HSZ = (size_t)Bb * Hh * Nn * HD;  // per q/k/v plane
  #pragma unroll
  for (int mt = 0; mt < 2; ++mt)
    #pragma unroll
    for (int nt = 0; nt < 4; ++nt)
      #pragma unroll
      for (int r = 0; r < 8; ++r) {
        int grow = rowBase + mt * 16 + r + 8 * hl;
        int gcol = colBase + nt * 16 + l16;
        float v = acc[mt][nt][r] + (bias ? bias[gcol] : 0.f);
        switch (epi) {
          case EPI_QKV: {   // q,k -> [B][H][N][hd] bf16 ; v -> transposed [B][H][hd][N]
            int which = gcol / Dd, ci = gcol - which * Dd;
            int hh = ci >> 6, dd = ci & 63;
            int bb = grow >> 11, pn = grow & (Nn - 1);
            if (which == 0) {
              v *= 0.125f;    // fold softmax scale hd^-0.5 into q
              outb[(((size_t)bb * Hh + hh) * Nn + pn) * HD + dd] = f2bf(v);
            } else if (which == 1) {
              outb[HSZ + (((size_t)bb * Hh + hh) * Nn + pn) * HD + dd] = f2bf(v);
            } else {          // V stored K-major so P*V B-fragments are contiguous
              outb[2 * HSZ + (((size_t)bb * Hh + hh) * HD + dd) * Nn + pn] = f2bf(v);
            }
          } break;
          case EPI_PROJ:                      // attn_out (+bias) -> left half of gi
            outb[(size_t)grow * (2 * Dd) + gcol] = f2bf(v);
            break;
          case EPI_GELU_BF16:
            outb[(size_t)grow * Nout + gcol] = f2bf(gelu_exact(v));
            break;
          case EPI_SIGMOID_F32:
            outf[(size_t)grow * Nout + gcol] = 1.f / (1.f + __expf(-v));
            break;
          case EPI_ADD_RES:
            outf[(size_t)grow * Nout + gcol] = aux[(size_t)grow * Nout + gcol] + v;
            break;
        }
      }
}

// ---------------------------------------- EdgeConv GEMM (gather fused in A)
// Virtual A row r=(b*K+kk)*N+n over gf[r][0:768] = concat(feat-xe, xe).
__global__ __launch_bounds__(256)
void knn_gemm_kernel(const unsigned short* __restrict__ nx,
                     const int* __restrict__ knn_ix,
                     const unsigned short* __restrict__ Bt,  // [384][768]
                     const float* __restrict__ bias,
                     float* __restrict__ knn_out) {
  const int tid = threadIdx.x;
  const int w = tid >> 5, lane = tid & 31, hl = lane >> 4, l16 = lane & 15;
  const int wm = w >> 1, wn = w & 1;
  const int rowBase = blockIdx.x * 128 + wm * 32;
  const int colBase = blockIdx.y * 128 + wn * 64;
  const int Kin = 2 * Dd;

  int idxr[2], xer[2];
  #pragma unroll
  for (int mt = 0; mt < 2; ++mt) {
    int gfrow = rowBase + mt * 16 + l16;
    idxr[mt] = knn_ix[gfrow];
    int bb = gfrow / (KNN * Nn);
    int pn = gfrow & (Nn - 1);
    xer[mt] = bb * Nn + pn;
  }

  v8f acc[2][4];
  #pragma unroll
  for (int i = 0; i < 2; ++i)
    #pragma unroll
    for (int j = 0; j < 4; ++j)
      #pragma unroll
      for (int e = 0; e < 8; ++e) acc[i][j][e] = 0.f;

  for (int kk = 0; kk < Kin; kk += 32) {
    const bool left = kk < Dd;   // first 384 cols: feat-xe ; last 384: xe
    Frag af[2], bf[4];
    #pragma unroll
    for (int mt = 0; mt < 2; ++mt) {
      int k0 = kk + hl * 8;
      if (left) {
        #pragma unroll
        for (int s = 0; s < 2; ++s) {
          int ko = k0 + s * 16;
          uint4 fa = *reinterpret_cast<const uint4*>(nx + (size_t)idxr[mt] * Dd + ko);
          uint4 fx = *reinterpret_cast<const uint4*>(nx + (size_t)xer[mt] * Dd + ko);
          af[mt].u4[s] = bfsub8(fa, fx);
        }
      } else {
        int ko = k0 - Dd;
        const unsigned short* p = nx + (size_t)xer[mt] * Dd + ko;
        af[mt].u4[0] = *reinterpret_cast<const uint4*>(p);
        af[mt].u4[1] = *reinterpret_cast<const uint4*>(p + 16);
      }
    }
    #pragma unroll
    for (int nt = 0; nt < 4; ++nt) {
      int col = colBase + nt * 16 + l16;
      int k0 = kk + hl * 8;
      const unsigned short* p = Bt + (size_t)col * Kin + k0;
      bf[nt].u4[0] = *reinterpret_cast<const uint4*>(p);
      bf[nt].u4[1] = *reinterpret_cast<const uint4*>(p + 16);
    }
    #pragma unroll
    for (int mt = 0; mt < 2; ++mt)
      #pragma unroll
      for (int nt = 0; nt < 4; ++nt)
        acc[mt][nt] = wmma_bf16(af[mt].v, bf[nt].v, acc[mt][nt]);
  }

  // epilogue: leaky_relu(0.2) then K-max-pool via float atomicMax
  #pragma unroll
  for (int mt = 0; mt < 2; ++mt)
    #pragma unroll
    for (int nt = 0; nt < 4; ++nt)
      #pragma unroll
      for (int r = 0; r < 8; ++r) {
        int grow = rowBase + mt * 16 + r + 8 * hl;
        int gcol = colBase + nt * 16 + l16;
        float v = acc[mt][nt][r] + bias[gcol];
        v = v > 0.f ? v : 0.2f * v;
        int bb = grow / (KNN * Nn);
        int pn = grow & (Nn - 1);
        atomicMax(knn_out + (size_t)(bb * Nn + pn) * Dd + gcol, v);
      }
}

// -------------------------------------------------- flash attention (bf16)
// grid (B*H, N/128); 8 waves; each wave: 16 query rows x 64-key blocks.
// K tile staged to LDS via Tensor Data Mover (one D# per wave, 8 rows each).
__global__ __launch_bounds__(256)
void attn_kernel(const unsigned short* __restrict__ qb,
                 const unsigned short* __restrict__ kb_,
                 const unsigned short* __restrict__ vT,   // [B][H][hd][N]
                 unsigned short* __restrict__ attn_cat) {
  __shared__ __align__(16) unsigned short lds_k[64 * 64];      // K tile (row-major)
  __shared__ __align__(16) unsigned short lds_p[8 * 16 * 64];  // per-wave P
  const int bh = blockIdx.x;
  const int bq = bh / Hh, hh = bh - bq * Hh;
  const int tid = threadIdx.x;
  const int w = tid >> 5, lane = tid & 31, hl = lane >> 4, l16 = lane & 15;
  const int q0 = blockIdx.y * 128 + w * 16;
  const unsigned short* qh = qb + (size_t)bh * Nn * HD;
  const unsigned short* kh = kb_ + (size_t)bh * Nn * HD;
  const unsigned short* vTh = vT + (size_t)bh * HD * Nn;

  // preload Q A-fragments (hd = 2 k-steps of 32)
  Frag qf[2];
  #pragma unroll
  for (int s = 0; s < 2; ++s) {
    int row = q0 + l16;
    int k0 = s * 32 + hl * 8;
    const unsigned short* p = qh + (size_t)row * HD + k0;
    qf[s].u4[0] = *reinterpret_cast<const uint4*>(p);
    qf[s].u4[1] = *reinterpret_cast<const uint4*>(p + 16);
  }

  v8f o[4];
  #pragma unroll
  for (int j = 0; j < 4; ++j)
    #pragma unroll
    for (int e = 0; e < 8; ++e) o[j][e] = 0.f;
  float mstate[8], lstate[8];
  #pragma unroll
  for (int r = 0; r < 8; ++r) { mstate[r] = -1e30f; lstate[r] = 0.f; }

  unsigned short* pw = lds_p + (size_t)w * 16 * 64;

  for (int kbk = 0; kbk < Nn / 64; ++kbk) {
    const int kRow0 = kbk * 64;
    __syncthreads();                       // protect previous K-tile reads
#if defined(HAVE_TDM)
    {
      // Tensor DMA: each wave moves its 8-row slice of the 64x64 bf16 K tile.
      unsigned long long ga =
          (unsigned long long)(uintptr_t)(kh + (size_t)(kRow0 + w * 8) * HD);
      unsigned int la =
          (unsigned int)(uintptr_t)lds_k + (unsigned int)(w * 8 * HD * 2);
      u32x4 g0 = { 1u,                                   // count=1 user descriptor
                   la,                                   // lds_addr
                   (unsigned int)(ga & 0xFFFFFFFFu),     // global_addr[31:0]
                   (unsigned int)((ga >> 32) & 0x1FFFFFFu) | 0x80000000u }; // type=2
      i32x8 g1 = { 0x00010000,          // data_size = 1 -> 2-byte elements
                   (int)(64u << 16),    // tensor_dim0 = 64
                   (int)(8u << 16),     // tensor_dim1 = 8 rows
                   (int)(64u << 16),    // tile_dim0 = 64
                   8,                   // tile_dim1 = 8
                   64,                  // tensor_dim0_stride = 64
                   0, 0 };
      i32x4 gz = { 0, 0, 0, 0 };
#if __clang_major__ >= 23
      i32x8 gz8 = { 0, 0, 0, 0, 0, 0, 0, 0 };
      __builtin_amdgcn_tensor_load_to_lds(g0, g1, gz, gz, gz8, 0);
#else
      __builtin_amdgcn_tensor_load_to_lds(g0, g1, gz, gz, 0);
#endif
      __builtin_amdgcn_s_wait_tensorcnt(0);
    }
#else
    // fallback: cooperative b128 copy of the K tile (2 chunks per thread)
    for (int c = tid; c < 512; c += 256) {
      int r = c >> 3, s = c & 7;
      *reinterpret_cast<uint4*>(lds_k + r * 64 + s * 8) =
          *reinterpret_cast<const uint4*>(kh + (size_t)(kRow0 + r) * HD + s * 8);
    }
#endif
    __syncthreads();

    // S = Q K^T : 4 key tiles of 16, 2 k-steps over hd, K from LDS
    v8f s4[4];
    #pragma unroll
    for (int j = 0; j < 4; ++j)
      #pragma unroll
      for (int e = 0; e < 8; ++e) s4[j][e] = 0.f;
    #pragma unroll
    for (int s = 0; s < 2; ++s)
      #pragma unroll
      for (int j = 0; j < 4; ++j) {
        Frag bfr;
        const unsigned short* p = lds_k + (j * 16 + l16) * 64 + s * 32 + hl * 8;
        bfr.u4[0] = *reinterpret_cast<const uint4*>(p);
        bfr.u4[1] = *reinterpret_cast<const uint4*>(p + 16);
        s4[j] = wmma_bf16(qf[s].v, bfr.v, s4[j]);
      }

    // online softmax per row (16-lane half-wave reductions)
    #pragma unroll
    for (int r = 0; r < 8; ++r) {
      float mx = fmaxf(fmaxf(s4[0][r], s4[1][r]), fmaxf(s4[2][r], s4[3][r]));
      #pragma unroll
      for (int off = 1; off < 16; off <<= 1) mx = fmaxf(mx, __shfl_xor(mx, off, 32));
      float mnew = fmaxf(mstate[r], mx);
      float corr = __expf(mstate[r] - mnew);
      float rs = 0.f;
      #pragma unroll
      for (int j = 0; j < 4; ++j) {
        float pv = __expf(s4[j][r] - mnew);
        s4[j][r] = pv;
        rs += pv;
      }
      #pragma unroll
      for (int off = 1; off < 16; off <<= 1) rs += __shfl_xor(rs, off, 32);
      lstate[r] = lstate[r] * corr + rs;
      mstate[r] = mnew;
      #pragma unroll
      for (int j = 0; j < 4; ++j) o[j][r] *= corr;
    }

    // P (C-layout f32) -> bf16 in wave-private LDS, reread as A-fragments
    #pragma unroll
    for (int j = 0; j < 4; ++j)
      #pragma unroll
      for (int r = 0; r < 8; ++r)
        pw[(r + 8 * hl) * 64 + j * 16 + l16] = f2bf(s4[j][r]);

    #pragma unroll
    for (int s = 0; s < 2; ++s) {          // key dim 64 = 2 k-steps
      Frag afr;
      int k0 = s * 32 + hl * 8;
      const unsigned short* p = pw + (size_t)l16 * 64 + k0;
      afr.u4[0] = *reinterpret_cast<const uint4*>(p);
      afr.u4[1] = *reinterpret_cast<const uint4*>(p + 16);
      #pragma unroll
      for (int j = 0; j < 4; ++j) {        // 4 hd tiles of 16; V^T direct from global
        Frag bfr;
        const unsigned short* q =
            vTh + (size_t)(j * 16 + l16) * Nn + kRow0 + s * 32 + hl * 8;
        bfr.u4[0] = *reinterpret_cast<const uint4*>(q);
        bfr.u4[1] = *reinterpret_cast<const uint4*>(q + 16);
        o[j] = wmma_bf16(afr.v, bfr.v, o[j]);
      }
    }
  }

  // normalize and scatter heads back into [B,N, H*hd] bf16
  #pragma unroll
  for (int j = 0; j < 4; ++j)
    #pragma unroll
    for (int r = 0; r < 8; ++r) {
      int row = q0 + r + 8 * hl;
      int col = j * 16 + l16;
      float v = o[j][r] / lstate[r];
      attn_cat[((size_t)bq * Nn + row) * Dd + hh * HD + col] = f2bf(v);
    }
}

// ------------------------------------------------ fuse gate + residual + LN2
__global__ __launch_bounds__(128)
void fused_ln2_kernel(const float* __restrict__ x, const float* __restrict__ gate,
                      const unsigned short* __restrict__ gi,
                      const float* __restrict__ knn_out,
                      const float* __restrict__ g, const float* __restrict__ b,
                      float* __restrict__ x1, unsigned short* __restrict__ ln2x) {
  __shared__ float red[128];
  int row = blockIdx.x, t = threadIdx.x;
  float vals[3];
  #pragma unroll
  for (int e = 0; e < 3; ++e) {
    int d = t + e * 128;
    float a  = bf2f(gi[(size_t)row * (2 * Dd) + d]);     // attn_out (bf16)
    float gt = gate[(size_t)row * Dd + d];
    float kn = knn_out[(size_t)row * Dd + d];
    float v  = x[(size_t)row * Dd + d] + (1.f - gt) * a + gt * kn;
    vals[e] = v;
    x1[(size_t)row * Dd + d] = v;
  }
  float mu = blockReduceSum128(vals[0] + vals[1] + vals[2], red) * (1.f / Dd);
  float d0 = vals[0] - mu, d1 = vals[1] - mu, d2 = vals[2] - mu;
  float var = blockReduceSum128(d0 * d0 + d1 * d1 + d2 * d2, red) * (1.f / Dd);
  float rstd = rsqrtf(var + 1e-5f);
  float dv[3] = { d0, d1, d2 };
  #pragma unroll
  for (int e = 0; e < 3; ++e) {
    int d = t + e * 128;
    ln2x[(size_t)row * Dd + d] = f2bf(dv[e] * rstd * g[d] + b[d]);
  }
}

// ============================================================== host driver
extern "C" void kernel_launch(void* const* d_in, const int* in_sizes, int n_in,
                              void* d_out, int out_size, void* d_ws, size_t ws_size,
                              hipStream_t stream) {
  (void)in_sizes; (void)n_in; (void)out_size; (void)ws_size;
  const float* x      = (const float*)d_in[0];
  const int*   knn_ix = (const int*)d_in[1];
  const float* ln1_g  = (const float*)d_in[2];
  const float* ln1_b  = (const float*)d_in[3];
  const float* w_qkv  = (const float*)d_in[4];
  const float* w_proj = (const float*)d_in[5];
  const float* b_proj = (const float*)d_in[6];
  const float* w_knn  = (const float*)d_in[7];
  const float* b_knn  = (const float*)d_in[8];
  const float* w_g1   = (const float*)d_in[9];
  const float* b_g1   = (const float*)d_in[10];
  const float* w_g2   = (const float*)d_in[11];
  const float* b_g2   = (const float*)d_in[12];
  const float* ln2_g  = (const float*)d_in[13];
  const float* ln2_b  = (const float*)d_in[14];
  const float* w_fc1  = (const float*)d_in[15];
  const float* b_fc1  = (const float*)d_in[16];
  const float* w_fc2  = (const float*)d_in[17];
  const float* b_fc2  = (const float*)d_in[18];
  float* out = (float*)d_out;

  char* ws = (char*)d_ws;
  size_t o = 0;
  auto alloc = [&](size_t bytes) { char* p = ws + o; o += (bytes + 255) & ~size_t(255); return p; };
  unsigned short* nx     = (unsigned short*)alloc((size_t)BN * Dd * 2);
  unsigned short* wqkvT  = (unsigned short*)alloc((size_t)Dd * 3 * Dd * 2);
  unsigned short* wprojT = (unsigned short*)alloc((size_t)Dd * Dd * 2);
  unsigned short* wknnT  = (unsigned short*)alloc((size_t)2 * Dd * Dd * 2);
  unsigned short* wg1T   = (unsigned short*)alloc((size_t)2 * Dd * Dd * 2);
  unsigned short* wg2T   = (unsigned short*)alloc((size_t)Dd * Dd * 2);
  unsigned short* wfc1T  = (unsigned short*)alloc((size_t)Dd * 4 * Dd * 2);
  unsigned short* wfc2T  = (unsigned short*)alloc((size_t)4 * Dd * Dd * 2);
  unsigned short* qkvb   = (unsigned short*)alloc((size_t)3 * Bb * Hh * Nn * HD * 2);
  unsigned short* attnc  = (unsigned short*)alloc((size_t)BN * Dd * 2);
  unsigned short* gi     = (unsigned short*)alloc((size_t)BN * 2 * Dd * 2);
  float*          knn_o  = (float*)alloc((size_t)BN * Dd * 4);
  unsigned short* ghid   = (unsigned short*)alloc((size_t)BN * Dd * 2);
  float*          gate   = (float*)alloc((size_t)BN * Dd * 4);
  float*          x1     = (float*)alloc((size_t)BN * Dd * 4);
  unsigned short* ln2x   = (unsigned short*)alloc((size_t)BN * Dd * 2);
  unsigned short* hbuf   = (unsigned short*)alloc((size_t)BN * 4 * Dd * 2);

  // 1) weight bf16 transposes  (Bt[N][K])
  auto cvt = [&](const float* src, unsigned short* dst, int R, int C) {
    int n = R * C;
    cvt_transpose_kernel<<<(n + 255) / 256, 256, 0, stream>>>(src, dst, R, C);
  };
  cvt(w_qkv,  wqkvT,  Dd,     3 * Dd);
  cvt(w_proj, wprojT, Dd,     Dd);
  cvt(w_knn,  wknnT,  2 * Dd, Dd);
  cvt(w_g1,   wg1T,   2 * Dd, Dd);
  cvt(w_g2,   wg2T,   Dd,     Dd);
  cvt(w_fc1,  wfc1T,  Dd,     4 * Dd);
  cvt(w_fc2,  wfc2T,  4 * Dd, Dd);

  // 2) LN1 -> nx (bf16)
  ln1_kernel<<<BN, 128, 0, stream>>>(x, ln1_g, ln1_b, nx);

  // 3) init knn max-pool accumulator
  {
    int n = BN * Dd;
    fill_neg_kernel<<<(n + 255) / 256, 256, 0, stream>>>(knn_o, n);
  }

  // 4) QKV projection (q scaled, k row-major, v transposed)
  gemm_bf16_kernel<<<dim3(BN / 128, (3 * Dd) / 128), 256, 0, stream>>>(
      Dd, 3 * Dd, nx, wqkvT, nullptr, EPI_QKV, nullptr, qkvb, nullptr);

  // 5) flash attention -> attn_cat bf16 [BN][D]
  {
    const size_t HSZ = (size_t)Bb * Hh * Nn * HD;
    attn_kernel<<<dim3(Bb * Hh, Nn / 128), 256, 0, stream>>>(
        qkvb, qkvb + HSZ, qkvb + 2 * HSZ, attnc);
  }

  // 6) output projection (+b_proj) -> left half of gi (bf16)
  gemm_bf16_kernel<<<dim3(BN / 128, Dd / 128), 256, 0, stream>>>(
      Dd, Dd, attnc, wprojT, b_proj, EPI_PROJ, nullptr, gi, nullptr);

  // 7) EdgeConv MLP with fused gather, leaky-relu, K-max-pool
  knn_gemm_kernel<<<dim3(GF_ROWS / 128, Dd / 128), 256, 0, stream>>>(
      nx, knn_ix, wknnT, b_knn, knn_o);

  // 8) knn_out -> right half of gi (bf16)
  {
    int n = BN * Dd;
    knn_to_gi_kernel<<<(n + 255) / 256, 256, 0, stream>>>(knn_o, gi, n);
  }

  // 9) gate MLP: gelu(gi@w_g1+b) -> ghid ; sigmoid(ghid@w_g2+b) -> gate
  gemm_bf16_kernel<<<dim3(BN / 128, Dd / 128), 256, 0, stream>>>(
      2 * Dd, Dd, gi, wg1T, b_g1, EPI_GELU_BF16, nullptr, ghid, nullptr);
  gemm_bf16_kernel<<<dim3(BN / 128, Dd / 128), 256, 0, stream>>>(
      Dd, Dd, ghid, wg2T, b_g2, EPI_SIGMOID_F32, gate, nullptr, nullptr);

  // 10) fused gate-combine + residual + LN2
  fused_ln2_kernel<<<BN, 128, 0, stream>>>(x, gate, gi, knn_o, ln2_g, ln2_b, x1, ln2x);

  // 11) FFN
  gemm_bf16_kernel<<<dim3(BN / 128, (4 * Dd) / 128), 256, 0, stream>>>(
      Dd, 4 * Dd, ln2x, wfc1T, b_fc1, EPI_GELU_BF16, nullptr, hbuf, nullptr);
  gemm_bf16_kernel<<<dim3(BN / 128, Dd / 128), 256, 0, stream>>>(
      4 * Dd, Dd, hbuf, wfc2T, b_fc2, EPI_ADD_RES, out, nullptr, x1);
}
